// Transformer_89120571392535
// MI455X (gfx1250) — matrix operand order repaired and checked by reading, not compiled, not run on hardware
//
#include <hip/hip_runtime.h>
#include <hip/hip_bf16.h>

typedef unsigned short ushort_t;
typedef __attribute__((ext_vector_type(16))) __bf16 v16bf;
typedef __attribute__((ext_vector_type(8)))  float  v8f;

union FragAB { v16bf v; uint4 u[2]; };

#define B_N    1024
#define HIST_N 199
#define D_N    256
#define HEADS  4
#define DH_N   64
#define DFF_N  1024
#define SK     224   // padded key length (7 * 32)
#define SQ     208   // padded query length (13 * 16)
#define NEG_INF_F (-4294967295.0f)   // -2^32 + 1, matches reference

static __device__ inline ushort_t f2bf(float x) {
    union { float f; unsigned int u; } c; c.f = x;
    unsigned int u = c.u + 0x7fffu + ((c.u >> 16) & 1u);   // RNE
    return (ushort_t)(u >> 16);
}
static __device__ inline float bf2f(ushort_t h) {
    union { unsigned int u; float f; } c; c.u = ((unsigned int)h) << 16;
    return c.f;
}

static __device__ inline v8f wmma_bf16(const FragAB& a, const FragAB& b, v8f c) {
    return __builtin_amdgcn_wmma_f32_16x16x32_bf16(false, a.v, false, b.v,
                                                   (short)0, c, false, false);
}

// ---------------------------------------------------------------------------
// Async global->LDS copy (CDNA5 GLOBAL_LOAD_ASYNC_TO_LDS_B128, ASYNCcnt).
// ldst low 32 bits of a flat shared pointer == LDS byte offset (ISA aperture
// rule: LDS address is the low 32 address bits).  nbytes must be a multiple of 16 and both
// addresses 16B-aligned.  Followed by s_wait_asynccnt 0; caller barriers.
// ---------------------------------------------------------------------------
static __device__ inline void async_stage(const void* __restrict__ gsrc,
                                          void* __restrict__ ldst,
                                          int nbytes, int tid) {
    unsigned lbase = (unsigned)(uintptr_t)ldst;
    const char* g  = (const char*)gsrc;
    for (int i = tid * 16; i < nbytes; i += 256 * 16) {
        asm volatile("global_load_async_to_lds_b128 %0, %1, off"
                     :: "v"(lbase + (unsigned)i), "v"(g + i) : "memory");
    }
    asm volatile("s_wait_asynccnt 0x0" ::: "memory");
}

// ---------------------------------------------------------------------------
// Weight cast kernels
// ---------------------------------------------------------------------------
__global__ void cast_copy_kernel(const float* __restrict__ src,
                                 ushort_t* __restrict__ dst, int n) {
    int i = blockIdx.x * blockDim.x + threadIdx.x;
    if (i < n) dst[i] = f2bf(src[i]);
}

// Qw/Kw/Vw [H][256][64] fp32 -> per-head transposed Wt[h][c][k] bf16
// (K-contiguous rows -> WMMA B-fragments are two 16B loads; also makes the
//  per-workgroup LDS staging a contiguous copy -> async-to-LDS eligible)
__global__ void cast_qkvT_kernel(const float* __restrict__ src,
                                 ushort_t* __restrict__ dst) {
    int i  = blockIdx.x * blockDim.x + threadIdx.x;  // over HEADS*DH_N*D_N
    int hh = i >> 14;          // head
    int c  = (i >> 8) & 63;    // dh column
    int k  = i & 255;          // d (reduction) index
    dst[i] = f2bf(src[hh * (D_N * DH_N) + k * DH_N + c]);
}

// W0 [256][1024] fp32 -> W0^T [1024][256] bf16 (K-contiguous for B-fragments)
__global__ void cast_w0t_kernel(const float* __restrict__ src,
                                ushort_t* __restrict__ dst) {
    int i = blockIdx.x * blockDim.x + threadIdx.x;   // i over 1024*256
    int n = i >> 8;       // dff index
    int k = i & 255;      // d index
    dst[i] = f2bf(src[k * DFF_N + n]);
}

// ---------------------------------------------------------------------------
// Attention kernel: one workgroup per (batch, head), 8 waves (wave32)
// ---------------------------------------------------------------------------
static __device__ inline void do_proj(const ushort_t* __restrict__ Xs,
                                      const ushort_t* __restrict__ Wst,
                                      ushort_t* __restrict__ dst,
                                      int nRowTiles, bool transposed, int dstStride,
                                      int wave, int lane) {
    int nT = nRowTiles * 4;                 // 4 column tiles of 16 (dh = 64)
    for (int t = wave; t < nT; t += 8) {
        int rt = t >> 2, ct = t & 3;
        v8f acc = {0.f,0.f,0.f,0.f,0.f,0.f,0.f,0.f};
#pragma unroll
        for (int ks = 0; ks < 8; ++ks) {    // K = 256 = 8 * 32
            FragAB A, Bw;
            int row = rt * 16 + (lane & 15);
            int ka  = ks * 32 + ((lane >> 4) << 4);
            A.u[0] = *(const uint4*)(Xs + row * D_N + ka);        // K e0..7
            A.u[1] = *(const uint4*)(Xs + row * D_N + ka + 8);   // K e8..15
            int col = ct * 16 + (lane & 15);
            int kb  = ks * 32 + ((lane >> 4) << 4);
            Bw.u[0] = *(const uint4*)(Wst + col * D_N + kb);      // 16 contiguous K
            Bw.u[1] = *(const uint4*)(Wst + col * D_N + kb + 8);
            acc = wmma_bf16(A, Bw, acc);
        }
#pragma unroll
        for (int e = 0; e < 8; ++e) {       // D layout: row = e + 8*half, col = lane&15
            int row = rt * 16 + e + ((lane >> 4) << 3);
            int col = ct * 16 + (lane & 15);
            if (transposed) dst[col * dstStride + row] = f2bf(acc[e]);
            else            dst[row * dstStride + col] = f2bf(acc[e]);
        }
    }
}

__global__ __launch_bounds__(256, 1)
void attn_kernel(const float* __restrict__ cand, const float* __restrict__ hist,
                 const int* __restrict__ lens,
                 const ushort_t* __restrict__ wQt, const ushort_t* __restrict__ wKt,
                 const ushort_t* __restrict__ wVt, ushort_t* __restrict__ Obuf) {
    extern __shared__ ushort_t smem[];
    ushort_t* Xs  = smem;                    // [224][256]
    ushort_t* Qh  = Xs  + SK * D_N;          // [208][64]
    ushort_t* Kh  = Qh  + SQ * DH_N;         // [224][64]
    ushort_t* VhT = Kh  + SK * DH_N;         // [64][224]  (V transposed)
    ushort_t* Wst = VhT + DH_N * SK;         // [64][256]  staged weight^T
    ushort_t* Pb  = Wst + DH_N * D_N;        // 8 waves * [16][224]

    const int b = blockIdx.x, h = blockIdx.y;
    const int tid = threadIdx.x, lane = tid & 31, wave = tid >> 5;

    // ---- stage Q weight^T for this head via async-to-LDS (contiguous bf16)
    async_stage(wQt + h * D_N * DH_N, Wst, D_N * DH_N * 2, tid);

    // ---- X[b] -> bf16 LDS (row 0 = candidate, rows 1..199 = history, pad 0)
    for (int i = tid; i < SK * D_N; i += 256) {
        int s = i >> 8, c = i & 255;
        float v = 0.f;
        if (s == 0)        v = cand[b * D_N + c];
        else if (s < 200)  v = hist[((s - 1) * B_N + b) * D_N + c];
        Xs[i] = f2bf(v);
    }
    __syncthreads();

    // ---- Q, K, V projections (V stored transposed for P@V B-fragments)
    do_proj(Xs, Wst, Qh, 13, false, DH_N, wave, lane); __syncthreads();
    async_stage(wKt + h * D_N * DH_N, Wst, D_N * DH_N * 2, tid); __syncthreads();
    do_proj(Xs, Wst, Kh, 14, false, DH_N, wave, lane); __syncthreads();
    async_stage(wVt + h * D_N * DH_N, Wst, D_N * DH_N * 2, tid); __syncthreads();
    do_proj(Xs, Wst, VhT, 14, true, SK, wave, lane); __syncthreads();

    const int len = lens[b] + 1;             // valid key positions: col < len
    ushort_t* Pw = Pb + wave * 16 * SK;      // per-wave P tile [16][224]

    for (int qt = wave; qt < 13; qt += 8) {
        // Q A-fragments for this 16-row query tile (dh = 64 -> 2 k-steps)
        FragAB aq0, aq1;
        {
            int row = qt * 16 + (lane & 15);
            int ka  = (lane >> 4) << 4;
            aq0.u[0] = *(const uint4*)(Qh + row * DH_N + ka);
            aq0.u[1] = *(const uint4*)(Qh + row * DH_N + ka + 8);
            aq1.u[0] = *(const uint4*)(Qh + row * DH_N + ka + 32);
            aq1.u[1] = *(const uint4*)(Qh + row * DH_N + ka + 40);
        }
        v8f sc[14];
#pragma unroll
        for (int kt = 0; kt < 14; ++kt) {
            FragAB bk0, bk1;
            int key = kt * 16 + (lane & 15);
            int kb  = (lane >> 4) << 4;
            // scores = Q @ K^T: B[kdim][key] = Kh[key][kdim] (row K-contiguous)
            bk0.u[0] = *(const uint4*)(Kh + key * DH_N + kb);
            bk0.u[1] = *(const uint4*)(Kh + key * DH_N + kb + 8);
            bk1.u[0] = *(const uint4*)(Kh + key * DH_N + kb + 32);
            bk1.u[1] = *(const uint4*)(Kh + key * DH_N + kb + 40);
            v8f d = {0.f,0.f,0.f,0.f,0.f,0.f,0.f,0.f};
            d = wmma_bf16(aq0, bk0, d);
            d = wmma_bf16(aq1, bk1, d);
            bool valid = (kt * 16 + (lane & 15)) < len;   // key col fixed per lane
#pragma unroll
            for (int e = 0; e < 8; ++e)
                sc[kt][e] = valid ? d[e] * 0.0625f : NEG_INF_F;  // /sqrt(256)
        }
        // ---- softmax: row r lives across 16 lanes of each half -> shfl_xor <=8
        float mx[8], sm[8];
#pragma unroll
        for (int e = 0; e < 8; ++e) {
            float m = -3.4e38f;
#pragma unroll
            for (int kt = 0; kt < 14; ++kt) m = fmaxf(m, sc[kt][e]);
            m = fmaxf(m, __shfl_xor(m, 1));
            m = fmaxf(m, __shfl_xor(m, 2));
            m = fmaxf(m, __shfl_xor(m, 4));
            m = fmaxf(m, __shfl_xor(m, 8));
            mx[e] = m;
        }
#pragma unroll
        for (int e = 0; e < 8; ++e) {
            float s = 0.f;
#pragma unroll
            for (int kt = 0; kt < 14; ++kt) {
                float p = __expf(sc[kt][e] - mx[e]);   // masked cols -> exp(-huge)=0
                sc[kt][e] = p; s += p;
            }
            s += __shfl_xor(s, 1);
            s += __shfl_xor(s, 2);
            s += __shfl_xor(s, 4);
            s += __shfl_xor(s, 8);
            sm[e] = 1.0f / s;
        }
        // ---- P -> bf16 LDS (per-wave, no barrier needed)
#pragma unroll
        for (int kt = 0; kt < 14; ++kt) {
            int col = kt * 16 + (lane & 15);
#pragma unroll
            for (int e = 0; e < 8; ++e) {
                int row = e + ((lane >> 4) << 3);
                Pw[row * SK + col] = f2bf(sc[kt][e] * sm[e]);
            }
        }
        // ---- O tile = P @ V  (K = 224 = 7 * 32)
#pragma unroll
        for (int nt = 0; nt < 4; ++nt) {
            v8f oac = {0.f,0.f,0.f,0.f,0.f,0.f,0.f,0.f};
#pragma unroll
            for (int ks = 0; ks < 7; ++ks) {
                FragAB a, bv;
                int row = lane & 15;
                int ka  = ks * 32 + ((lane >> 4) << 4);
                a.u[0] = *(const uint4*)(Pw + row * SK + ka);
                a.u[1] = *(const uint4*)(Pw + row * SK + ka + 8);
                int n  = nt * 16 + (lane & 15);
                int kb = ks * 32 + ((lane >> 4) << 4);
                bv.u[0] = *(const uint4*)(VhT + n * SK + kb);
                bv.u[1] = *(const uint4*)(VhT + n * SK + kb + 8);
                oac = wmma_bf16(a, bv, oac);
            }
#pragma unroll
            for (int e = 0; e < 8; ++e) {
                int row = qt * 16 + e + ((lane >> 4) << 3);
                int col = h * DH_N + nt * 16 + (lane & 15);
                Obuf[(b * SQ + row) * D_N + col] = f2bf(oac[e]);
            }
        }
    }
}

// ---------------------------------------------------------------------------
// FFN: T = O@W0, h = relu(T+b0), hsum = sum_s h, out = hsum@W1 + 200*b1
// ---------------------------------------------------------------------------
__global__ __launch_bounds__(256, 1)
void ffn_kernel(const ushort_t* __restrict__ Obuf, const ushort_t* __restrict__ w0T,
                const float* __restrict__ b0, const ushort_t* __restrict__ w1,
                const float* __restrict__ b1, float* __restrict__ out) {
    extern __shared__ ushort_t smem[];
    ushort_t* Os  = smem;                       // [208][256] bf16
    float*   hsum = (float*)(smem + SQ * D_N);  // [1024]
    const int b = blockIdx.x, tid = threadIdx.x, lane = tid & 31, wave = tid >> 5;

    // load O[b] via async global->LDS (pure bf16 copy, 106496 B)
    async_stage(Obuf + (size_t)b * SQ * D_N, Os, SQ * D_N * 2, tid);
    __syncthreads();

    for (int ct = wave; ct < DFF_N / 16; ct += 8) {   // 64 column tiles, wave-exclusive
        FragAB bw[8];
        int n = ct * 16 + (lane & 15);
#pragma unroll
        for (int ks = 0; ks < 8; ++ks) {              // B-frags from W0^T (global)
            int kb = ks * 32 + ((lane >> 4) << 4);
            bw[ks].u[0] = *(const uint4*)(w0T + n * D_N + kb);
            bw[ks].u[1] = *(const uint4*)(w0T + n * D_N + kb + 8);
        }
        float bias = b0[n];
        float colacc = 0.f;
        for (int rt = 0; rt < 13; ++rt) {
            v8f acc = {0.f,0.f,0.f,0.f,0.f,0.f,0.f,0.f};
#pragma unroll
            for (int ks = 0; ks < 8; ++ks) {
                FragAB a;
                int row = rt * 16 + (lane & 15);
                int ka  = ks * 32 + ((lane >> 4) << 4);
                a.u[0] = *(const uint4*)(Os + row * D_N + ka);
                a.u[1] = *(const uint4*)(Os + row * D_N + ka + 8);
                acc = wmma_bf16(a, bw[ks], acc);
            }
#pragma unroll
            for (int e = 0; e < 8; ++e) {
                int grow = rt * 16 + e + ((lane >> 4) << 3);
                if (grow < 200) colacc += fmaxf(acc[e] + bias, 0.f);  // relu(T+b0)
            }
        }
        colacc += __shfl_xor(colacc, 16);             // combine the two lane halves
        if (lane < 16) hsum[ct * 16 + lane] = colacc; // wave-exclusive column tile
    }
    __syncthreads();

    // GEMV: out[b][d] = sum_f hsum[f]*W1[f][d] + 200*b1[d]
    int d = tid;
    float acc = 0.f;
    for (int f = 0; f < DFF_N; f += 4) {
        acc += hsum[f + 0] * bf2f(w1[(f + 0) * D_N + d]);
        acc += hsum[f + 1] * bf2f(w1[(f + 1) * D_N + d]);
        acc += hsum[f + 2] * bf2f(w1[(f + 2) * D_N + d]);
        acc += hsum[f + 3] * bf2f(w1[(f + 3) * D_N + d]);
    }
    out[b * D_N + d] = acc + 200.0f * b1[d];
}

// ---------------------------------------------------------------------------
extern "C" void kernel_launch(void* const* d_in, const int* in_sizes, int n_in,
                              void* d_out, int out_size, void* d_ws, size_t ws_size,
                              hipStream_t stream) {
    const float* cand = (const float*)d_in[0];
    const float* hist = (const float*)d_in[1];
    const int*   lens = (const int*)d_in[2];
    const float* Qw   = (const float*)d_in[3];
    const float* Kw   = (const float*)d_in[4];
    const float* Vw   = (const float*)d_in[5];
    const float* W0   = (const float*)d_in[6];
    const float* b0   = (const float*)d_in[7];
    const float* W1   = (const float*)d_in[8];
    const float* b1   = (const float*)d_in[9];
    float* out = (float*)d_out;

    // workspace carve (bytes, 256-aligned)
    char* ws = (char*)d_ws;
    const size_t QKV_B = (size_t)HEADS * D_N * DH_N * 2;   // 131072 each
    ushort_t* wQt = (ushort_t*)(ws);
    ushort_t* wKt = (ushort_t*)(ws + QKV_B);
    ushort_t* wVt = (ushort_t*)(ws + 2 * QKV_B);
    ushort_t* w0T = (ushort_t*)(ws + 3 * QKV_B);                        // 524288 B
    ushort_t* w1b = (ushort_t*)(ws + 3 * QKV_B + 524288);               // 524288 B
    ushort_t* Ob  = (ushort_t*)(ws + 3 * QKV_B + 2 * 524288);           // B*208*256*2

    const int nqkv = HEADS * D_N * DH_N;       // 65536
    cast_qkvT_kernel<<<nqkv / 256, 256, 0, stream>>>(Qw, wQt);
    cast_qkvT_kernel<<<nqkv / 256, 256, 0, stream>>>(Kw, wKt);
    cast_qkvT_kernel<<<nqkv / 256, 256, 0, stream>>>(Vw, wVt);
    cast_w0t_kernel<<<(D_N * DFF_N) / 256, 256, 0, stream>>>(W0, w0T);
    cast_copy_kernel<<<(DFF_N * D_N) / 256, 256, 0, stream>>>(W1, w1b, DFF_N * D_N);

    const size_t ATTN_SMEM =
        (size_t)(SK * D_N + SQ * DH_N + SK * DH_N + DH_N * SK + DH_N * D_N +
                 8 * 16 * SK) * 2;             // 288768 B (< 320 KB/WGP)
    attn_kernel<<<dim3(B_N, HEADS), 256, ATTN_SMEM, stream>>>(
        cand, hist, lens, wQt, wKt, wVt, Ob);

    const size_t FFN_SMEM = (size_t)SQ * D_N * 2 + DFF_N * 4;   // 110592 B
    ffn_kernel<<<B_N, 256, FFN_SMEM, stream>>>(Ob, w0T, b0, w1b, b1, out);
}
